// PointRendSemSegHead_80848464379913
// MI455X (gfx1250) — matrix operand order repaired and checked by reading, not compiled, not run on hardware
//
#include <hip/hip_runtime.h>
#include <stdint.h>

// ---------------- problem constants ----------------
#define NB    8
#define CCH   19
#define HC    64
#define WC    64
#define NIN   256
#define HF    128
#define WF    128
#define FIN   275
#define KPAD  288
#define KDW   144          // dwords per padded activation row (288 bf16)
#define PNT   16384
#define SS    49152        // NUM_POINTS * OVERSAMPLE
#define NIMP  12288        // int(0.75 * 16384)
#define NRAND 4096
#define LSTR  20           // LDS tile row stride in dwords (bank-conflict-free, 16B aligned)

// output regions (floats)
#define OUT_COARSE_ELTS  (NB*CCH*HC*WC)          // 622592
#define OUT_LOGITS_ELTS  (NB*CCH*PNT)            // 2490368

// ---------------- CDNA5 async global->LDS path (guarded) ----------------
#define AS1 __attribute__((address_space(1)))
#define AS3 __attribute__((address_space(3)))
typedef int v4i_ls __attribute__((vector_size(16)));   // matches builtin param pointee

#if defined(__AMDGCN__) && __has_builtin(__builtin_amdgcn_global_load_async_to_lds_b128)
#define HAVE_ASYNC_LDS 1
#define ASYNC_B128(g, l, off) \
    __builtin_amdgcn_global_load_async_to_lds_b128((AS1 v4i_ls*)(g), (AS3 v4i_ls*)(l), (off), 0)
#else
#define HAVE_ASYNC_LDS 0
#endif

#if defined(__AMDGCN__) && __has_builtin(__builtin_amdgcn_s_wait_asynccnt)
#define WAIT_ASYNC(n) __builtin_amdgcn_s_wait_asynccnt(n)
#elif defined(__AMDGCN__)
#define WAIT_ASYNC(n) asm volatile("s_wait_asynccnt %0" :: "i"(n) : "memory")
#else
#define WAIT_ASYNC(n)
#endif

// ---------------- WMMA types ----------------
typedef __attribute__((ext_vector_type(16))) __bf16 v16bf;
typedef __attribute__((ext_vector_type(8)))  float  v8f;
union Frag { unsigned u[8]; v16bf v; };

// ---------------- helpers ----------------
__device__ __forceinline__ unsigned pcg_hash(unsigned v) {
    v = v * 747796405u + 2891336453u;
    unsigned w = ((v >> ((v >> 28u) + 4u)) ^ v) * 277803737u;
    return (w >> 22u) ^ w;
}
__device__ __forceinline__ float u01(unsigned h) {
    return (float)(h >> 8) * (1.0f / 16777216.0f);
}
__device__ __forceinline__ unsigned f2bf(float f) {   // RNE f32 -> bf16 bits
    unsigned x = __float_as_uint(f);
    return (x + 0x7FFFu + ((x >> 16) & 1u)) >> 16;
}
__device__ __forceinline__ float bilin(const float* __restrict__ f,
                                       float cx, float cy, int H, int W) {
    float x = cx * (float)W - 0.5f, y = cy * (float)H - 0.5f;
    float xf = floorf(x), yf = floorf(y);
    float wx1 = x - xf, wy1 = y - yf;
    int x0 = (int)xf, y0 = (int)yf, x1 = x0 + 1, y1 = y0 + 1;
    float vx0 = (x0 >= 0 && x0 < W) ? 1.f : 0.f;
    float vx1 = (x1 >= 0 && x1 < W) ? 1.f : 0.f;
    float vy0 = (y0 >= 0 && y0 < H) ? 1.f : 0.f;
    float vy1 = (y1 >= 0 && y1 < H) ? 1.f : 0.f;
    int a0 = min(max(x0, 0), W - 1), a1 = min(max(x1, 0), W - 1);
    int b0 = min(max(y0, 0), H - 1), b1 = min(max(y1, 0), H - 1);
    float s = 0.f;
    s += (1.f - wy1) * (1.f - wx1) * vy0 * vx0 * f[b0 * W + a0];
    s += (1.f - wy1) * wx1         * vy0 * vx1 * f[b0 * W + a1];
    s += wy1 * (1.f - wx1)         * vy1 * vx0 * f[b1 * W + a0];
    s += wy1 * wx1                 * vy1 * vx1 * f[b1 * W + a1];
    return s;
}

// ---------------- stage 0: weight convert (f32 -> padded bf16) ----------------
__global__ void cvt_w_kernel(const float* __restrict__ w, uint16_t* __restrict__ wb,
                             int O, int Opad) {
    int idx = blockIdx.x * blockDim.x + threadIdx.x;
    if (idx >= Opad * KPAD) return;
    int o = idx / KPAD, k = idx % KPAD;
    float v = (o < O && k < FIN) ? w[(size_t)o * FIN + k] : 0.f;
    wb[idx] = (uint16_t)f2bf(v);
}

// ---------------- stage 1: candidate RNG ----------------
__global__ void gen_cand_kernel(float* __restrict__ cand) {
    int i = blockIdx.x * blockDim.x + threadIdx.x;
    if (i >= NB * SS * 2) return;
    cand[i] = u01(pcg_hash(0x12345u ^ (unsigned)i));
}

// ---------------- stage 2: coarse sample + uncertainty key ----------------
__global__ void sample_unc_kernel(const float* __restrict__ coarse,
                                  const float* __restrict__ cand,
                                  unsigned* __restrict__ key) {
    int i = blockIdx.x * blockDim.x + threadIdx.x;
    if (i >= NB * SS) return;
    int n = i / SS;
    float cx = cand[2 * (size_t)i], cy = cand[2 * (size_t)i + 1];
    const float* base = coarse + (size_t)n * CCH * HC * WC;
    float m1 = -3.4e38f, m2 = -3.4e38f;
    for (int c = 0; c < CCH; ++c) {
        float v = bilin(base + (size_t)c * HC * WC, cx, cy, HC, WC);
        if (v > m1) { m2 = m1; m1 = v; } else if (v > m2) m2 = v;
    }
    float unc = m2 - m1;                 // <= 0
    unsigned b = __float_as_uint(unc);
    key[i] = b ^ ((b & 0x80000000u) ? 0xFFFFFFFFu : 0x80000000u);
}

// ---------------- stage 3: radix-select top NIMP ----------------
__global__ void hist_kernel(const unsigned* __restrict__ key,
                            unsigned* __restrict__ hist) {
    int i = blockIdx.x * blockDim.x + threadIdx.x;
    if (i >= NB * SS) return;
    int n = i / SS;
    atomicAdd(&hist[n * 1024 + (key[i] >> 22)], 1u);
}
__global__ void scan_kernel(const unsigned* __restrict__ hist, int* __restrict__ meta) {
    int n = threadIdx.x;
    if (n >= NB) return;
    long cum = 0; int cut = 0;
    for (int b = 1023; b >= 0; --b) {
        unsigned h = hist[n * 1024 + b];
        if (cum + (long)h >= (long)NIMP) { cut = b; break; }
        cum += h;
    }
    meta[4 * n] = cut; meta[4 * n + 1] = (int)cum;
    meta[4 * n + 2] = NIMP - (int)cum; meta[4 * n + 3] = 0;
}
__global__ void select_kernel(const float* __restrict__ cand,
                              const unsigned* __restrict__ key,
                              const int* __restrict__ meta,
                              unsigned* __restrict__ cnt,
                              float* __restrict__ pc) {
    int i = blockIdx.x * blockDim.x + threadIdx.x;
    if (i >= NB * SS) return;
    int n = i / SS;
    int cut = meta[4 * n], hi = meta[4 * n + 1], rem = meta[4 * n + 2];
    int bin = (int)(key[i] >> 22);
    int slot = -1;
    if (bin > cut) {
        slot = (int)atomicAdd(&cnt[2 * n], 1u);
    } else if (bin == cut) {
        unsigned j = atomicAdd(&cnt[2 * n + 1], 1u);
        if ((int)j < rem) slot = hi + (int)j;
    }
    if (slot >= 0) {
        size_t d = ((size_t)n * PNT + slot) * 2;
        pc[d]     = cand[2 * (size_t)i];
        pc[d + 1] = cand[2 * (size_t)i + 1];
    }
}
__global__ void rand_fill_kernel(float* __restrict__ pc) {
    int i = blockIdx.x * blockDim.x + threadIdx.x;
    if (i >= NB * NRAND) return;
    int n = i / NRAND, r = i % NRAND;
    size_t d = ((size_t)n * PNT + NIMP + r) * 2;
    pc[d]     = u01(pcg_hash(0xDEADBEEFu ^ (unsigned)(2 * i)));
    pc[d + 1] = u01(pcg_hash(0xDEADBEEFu ^ (unsigned)(2 * i + 1)));
}

// ---------------- stage 4: build X0 = concat(fine, coarse) as bf16 (N,P,288) ----------------
__global__ __launch_bounds__(288) void build_x0_kernel(
    const float* __restrict__ feat, const float* __restrict__ coarse,
    const float* __restrict__ pc, uint16_t* __restrict__ x0) {
    int idx = blockIdx.x;              // n*P + p
    int c = threadIdx.x;               // 0..287
    int n = idx / PNT;
    float cx = pc[2 * (size_t)idx], cy = pc[2 * (size_t)idx + 1];
    float v = 0.f;
    if (c < NIN)
        v = bilin(feat + ((size_t)n * NIN + c) * HF * WF, cx, cy, HF, WF);
    else if (c < FIN)
        v = bilin(coarse + ((size_t)n * CCH + (c - NIN)) * HC * WC, cx, cy, HC, WC);
    x0[(size_t)idx * KPAD + c] = (uint16_t)f2bf(v);
}

// ---------------- stage 5: hidden GEMM (O=256, K=288, relu, bf16 out) ----------------
// Tile: 128 points x 128 outputs per block; 8 waves, each 16 outputs x 128 points.
// X tile staged global->LDS; with CDNA5 async-LDS loads, double-buffered so the
// DMA of tile k+1 overlaps the 8 WMMAs of tile k (ASYNCcnt retires in order,
// so waiting <=2 guarantees the older tile's 2 loads per thread are resident).
__global__ __launch_bounds__(256) void gemm_hidden_kernel(
    const uint32_t* __restrict__ xin, const uint16_t* __restrict__ wb,
    const float* __restrict__ bias, uint32_t* __restrict__ xout) {
    __shared__ uint32_t lds[2][128 * LSTR];
    const int tid = threadIdx.x;
    const int lane = tid & 31, wave = tid >> 5;
    const int half = lane >> 4, row = lane & 15;
    const int n = blockIdx.z;
    const int p0 = blockIdx.x * 128;
    const int o_base = blockIdx.y * 128 + wave * 16;
    const uint32_t* wdw = (const uint32_t*)wb;

    v8f acc[8];
#pragma unroll
    for (int t = 0; t < 8; ++t) acc[t] = (v8f){0.f,0.f,0.f,0.f,0.f,0.f,0.f,0.f};

    const size_t xrow0 = ((size_t)n * PNT + p0) * KDW;
    // this thread's staging slice: point p = tid>>1, K-half h = tid&1 (8 dwords)
    const int sp = tid >> 1, sh = tid & 1;
    const uint32_t* gbase = xin + xrow0 + (size_t)sp * KDW + sh * 8;
    uint32_t* lb[2] = { &lds[0][sp * LSTR + sh * 8], &lds[1][sp * LSTR + sh * 8] };

#if HAVE_ASYNC_LDS
    ASYNC_B128(gbase, lb[0], 0);
    ASYNC_B128(gbase, lb[0], 16);
#endif
    for (int kt = 0; kt < 9; ++kt) {
#if HAVE_ASYNC_LDS
        if (kt < 8) {
            const uint32_t* g = gbase + (kt + 1) * 16;
            ASYNC_B128(g, lb[(kt + 1) & 1], 0);
            ASYNC_B128(g, lb[(kt + 1) & 1], 16);
            WAIT_ASYNC(2);
        } else {
            WAIT_ASYNC(0);
        }
        __syncthreads();
        const uint32_t* bufp = &lds[kt & 1][0];
#else
#pragma unroll
        for (int r2 = 0; r2 < 8; ++r2) {
            int idx = tid + r2 * 256;
            int p = idx >> 4, kd = idx & 15;
            lds[0][p * LSTR + kd] = xin[xrow0 + (size_t)p * KDW + kt * 16 + kd];
        }
        __syncthreads();
        const uint32_t* bufp = &lds[0][0];
#endif
        Frag a;
        const uint32_t* wrow = wdw + (size_t)(o_base + row) * KDW + kt * 16;
        if (kt < 8) __builtin_prefetch(wrow + 16, 0, 1);
#pragma unroll
        for (int j = 0; j < 8; ++j) {
            int pairk = j + half * 8;   // A-frag K striping
            a.u[j] = wrow[pairk];
        }
        // t = 0 (no reuse hint), t = 1..7 reuse the A operand (identical wmma)
        {
            Frag b;
#pragma unroll
            for (int j = 0; j < 8; ++j) b.u[j] = bufp[row * LSTR + half * 8 + j];
            acc[0] = __builtin_amdgcn_wmma_f32_16x16x32_bf16(
                false, a.v, false, b.v, (short)0, acc[0], false, false);
        }
#pragma unroll
        for (int t = 1; t < 8; ++t) {
            Frag b;
            int pcol = t * 16 + row;
#pragma unroll
            for (int j = 0; j < 8; ++j) b.u[j] = bufp[pcol * LSTR + half * 8 + j];
            acc[t] = __builtin_amdgcn_wmma_f32_16x16x32_bf16(
                false, a.v, false, b.v, (short)0, acc[t], true, false);
        }
        __syncthreads();
    }
    // epilogue: bias + relu + bf16 pack
#pragma unroll
    for (int t = 0; t < 8; ++t) {
        int p = p0 + t * 16 + row;
        size_t rb = ((size_t)n * PNT + p) * KDW;
        int od = o_base + half * 8;
#pragma unroll
        for (int q = 0; q < 4; ++q) {
            float v0 = fmaxf(acc[t][2 * q]     + bias[od + 2 * q],     0.f);
            float v1 = fmaxf(acc[t][2 * q + 1] + bias[od + 2 * q + 1], 0.f);
            xout[rb + od / 2 + q] = f2bf(v0) | (f2bf(v1) << 16);
        }
    }
    // carry coarse channels (dwords 128..143 = channels 256..287) through
    if (blockIdx.y == 0) {
        for (int idx = tid; idx < 2048; idx += 256) {
            int p = idx >> 4, d = idx & 15;
            size_t rb = ((size_t)n * PNT + p0 + p) * KDW;
            xout[rb + 128 + d] = xin[rb + 128 + d];
        }
    }
}

// ---------------- stage 6: final GEMM (O=19 padded to 32, f32 out (N,C,P)) ----------------
__global__ __launch_bounds__(256) void gemm_final_kernel(
    const uint32_t* __restrict__ xin, const uint16_t* __restrict__ wb,
    const float* __restrict__ bias, float* __restrict__ out) {
    __shared__ uint32_t lds[128 * LSTR];
    const int tid = threadIdx.x;
    const int lane = tid & 31, wave = tid >> 5;
    const int half = lane >> 4, row = lane & 15;
    const int n = blockIdx.z;
    const int p0 = blockIdx.x * 128;
    const uint32_t* wdw = (const uint32_t*)wb;

    v8f acc0 = (v8f){0.f,0.f,0.f,0.f,0.f,0.f,0.f,0.f};
    v8f acc1 = (v8f){0.f,0.f,0.f,0.f,0.f,0.f,0.f,0.f};

    const size_t xrow0 = ((size_t)n * PNT + p0) * KDW;
    for (int kt = 0; kt < 9; ++kt) {
#pragma unroll
        for (int r2 = 0; r2 < 8; ++r2) {
            int idx = tid + r2 * 256;
            int p = idx >> 4, kd = idx & 15;
            lds[p * LSTR + kd] = xin[xrow0 + (size_t)p * KDW + kt * 16 + kd];
        }
        __syncthreads();
        Frag a0, a1, b;
#pragma unroll
        for (int j = 0; j < 8; ++j) {
            int pairk = j + half * 8;
            a0.u[j] = wdw[(size_t)(row)      * KDW + kt * 16 + pairk];
            a1.u[j] = wdw[(size_t)(16 + row) * KDW + kt * 16 + pairk];
        }
        int pcol = wave * 16 + row;
#pragma unroll
        for (int j = 0; j < 8; ++j) b.u[j] = lds[pcol * LSTR + half * 8 + j];
        acc0 = __builtin_amdgcn_wmma_f32_16x16x32_bf16(
            false, a0.v, false, b.v, (short)0, acc0, false, false);
        acc1 = __builtin_amdgcn_wmma_f32_16x16x32_bf16(
            false, a1.v, false, b.v, (short)0, acc1, false, true);
        __syncthreads();
    }
    int p = p0 + wave * 16 + row;
#pragma unroll
    for (int r = 0; r < 8; ++r) {
        int o = half * 8 + r;
        if (o < CCH) out[((size_t)n * CCH + o) * PNT + p] = acc0[r] + bias[o];
        int o2 = 16 + half * 8 + r;
        if (o2 < CCH) out[((size_t)n * CCH + o2) * PNT + p] = acc1[r] + bias[o2];
    }
}

// ---------------- launch ----------------
extern "C" void kernel_launch(void* const* d_in, const int* in_sizes, int n_in,
                              void* d_out, int out_size, void* d_ws, size_t ws_size,
                              hipStream_t stream) {
    (void)in_sizes; (void)n_in; (void)out_size; (void)ws_size;
    const float* coarse = (const float*)d_in[0];
    const float* feats  = (const float*)d_in[1];
    const float* w0 = (const float*)d_in[2]; const float* b0 = (const float*)d_in[3];
    const float* w1 = (const float*)d_in[4]; const float* b1 = (const float*)d_in[5];
    const float* w2 = (const float*)d_in[6]; const float* b2 = (const float*)d_in[7];
    const float* wp = (const float*)d_in[8]; const float* bp = (const float*)d_in[9];

    char* ws = (char*)d_ws;
    uint16_t* w0b = (uint16_t*)(ws + 0);          // 256*288*2 = 147456
    uint16_t* w1b = (uint16_t*)(ws + 147456);
    uint16_t* w2b = (uint16_t*)(ws + 294912);
    uint16_t* wpb = (uint16_t*)(ws + 442368);     // 32*288*2 = 18432
    float*    cand = (float*)(ws + 460800);       // 8*49152*2*4 = 3145728
    unsigned* key  = (unsigned*)(ws + 3606528);   // 8*49152*4 = 1572864
    unsigned* hist = (unsigned*)(ws + 5179392);   // 8*1024*4 = 32768
    int*      meta = (int*)(ws + 5212160);        // 8*4*4 = 128
    unsigned* cnt  = (unsigned*)(ws + 5212288);   // 8*2*4 = 64
    uint16_t* x0   = (uint16_t*)(ws + 6291456);   // 75497472 bytes
    uint16_t* x1   = (uint16_t*)(ws + 81788928);  // 75497472 bytes (end 157286400)

    float* out        = (float*)d_out;
    float* out_logits = out + OUT_COARSE_ELTS;
    float* pc         = out + OUT_COARSE_ELTS + OUT_LOGITS_ELTS;

    // region 0: pass-through coarse logits
    (void)hipMemcpyAsync(out, coarse, (size_t)OUT_COARSE_ELTS * sizeof(float),
                         hipMemcpyDeviceToDevice, stream);
    (void)hipMemsetAsync(hist, 0, NB * 1024 * sizeof(unsigned), stream);
    (void)hipMemsetAsync(cnt,  0, NB * 2 * sizeof(unsigned), stream);

    cvt_w_kernel<<<(256 * KPAD + 255) / 256, 256, 0, stream>>>(w0, w0b, 256, 256);
    cvt_w_kernel<<<(256 * KPAD + 255) / 256, 256, 0, stream>>>(w1, w1b, 256, 256);
    cvt_w_kernel<<<(256 * KPAD + 255) / 256, 256, 0, stream>>>(w2, w2b, 256, 256);
    cvt_w_kernel<<<(32  * KPAD + 255) / 256, 256, 0, stream>>>(wp, wpb, CCH, 32);

    gen_cand_kernel<<<(NB * SS * 2 + 255) / 256, 256, 0, stream>>>(cand);
    sample_unc_kernel<<<(NB * SS + 255) / 256, 256, 0, stream>>>(coarse, cand, key);
    hist_kernel<<<(NB * SS + 255) / 256, 256, 0, stream>>>(key, hist);
    scan_kernel<<<1, 32, 0, stream>>>(hist, meta);
    select_kernel<<<(NB * SS + 255) / 256, 256, 0, stream>>>(cand, key, meta, cnt, pc);
    rand_fill_kernel<<<(NB * NRAND + 255) / 256, 256, 0, stream>>>(pc);

    build_x0_kernel<<<NB * PNT, 288, 0, stream>>>(feats, coarse, pc, x0);

    dim3 gh(PNT / 128, 2, NB);
    gemm_hidden_kernel<<<gh, 256, 0, stream>>>((const uint32_t*)x0, w0b, b0, (uint32_t*)x1);
    gemm_hidden_kernel<<<gh, 256, 0, stream>>>((const uint32_t*)x1, w1b, b1, (uint32_t*)x0);
    gemm_hidden_kernel<<<gh, 256, 0, stream>>>((const uint32_t*)x0, w2b, b2, (uint32_t*)x1);

    dim3 gf(PNT / 128, 1, NB);
    gemm_final_kernel<<<gf, 256, 0, stream>>>((const uint32_t*)x1, wpb, bp, out_logits);
}
